// SpatialPyramidAggregator_85813446574844
// MI455X (gfx1250) — matrix operand order repaired and checked by reading, not compiled, not run on hardware
//
#include <hip/hip_runtime.h>
#include <hip/hip_bf16.h>

typedef __attribute__((ext_vector_type(16))) __bf16 v16bf;
typedef __attribute__((ext_vector_type(8)))  __bf16 bf16x8;
typedef __attribute__((ext_vector_type(8)))  float  v8f;
typedef __attribute__((ext_vector_type(4)))  float  f32x4;
typedef __attribute__((ext_vector_type(4)))  int    v4i;
// pointer-to-AS1/AS3 int4, matching the async-to-LDS builtin's parameter types
typedef __attribute__((address_space(1))) v4i* as1_v4i;
typedef __attribute__((address_space(3))) v4i* as3_v4i;

#define BATCH 64
#define CDIM  64
#define KDIM  12864   // 201 * 64
#define HID   256
#define FINEB 128     // finest level (4,4,8) bins per batch

#if defined(__HIP_DEVICE_COMPILE__) && \
    __has_builtin(__builtin_amdgcn_global_load_async_to_lds_b128) && \
    __has_builtin(__builtin_amdgcn_s_wait_asynccnt)
#define HAVE_ASYNC_LDS 1
#else
#define HAVE_ASYNC_LDS 0
#endif

// ---- order-preserving float <-> uint key for atomicMax-based segment max ----
__device__ __forceinline__ unsigned fkey(float f) {
  unsigned u = __float_as_uint(f);
  return (u & 0x80000000u) ? ~u : (u | 0x80000000u);
}
__device__ __forceinline__ float fdec(unsigned k) {
  unsigned u = (k & 0x80000000u) ? (k & 0x7FFFFFFFu) : ~k;
  return __uint_as_float(u);
}

// ---------------- workspace zeroing (accumulators must start at 0) ----------
__global__ void k_zero(unsigned* __restrict__ p, int nwords) {
  int i = blockIdx.x * blockDim.x + threadIdx.x;
  if (i < nwords) p[i] = 0u;
}

// ---------------- fp32 -> bf16 weight conversion ----------------------------
__global__ void k_cvt(const float* __restrict__ w1, const float* __restrict__ w2,
                      __bf16* __restrict__ o1, __bf16* __restrict__ o2,
                      int n1, int n2) {
  int i = blockIdx.x * blockDim.x + threadIdx.x;
  if (i < n1) o1[i] = (__bf16)w1[i];
  if (i < n2) o2[i] = (__bf16)w2[i];
}

// ---------------- attention MLP via WMMA: raw scores + per-batch max --------
// each wave handles 4 consecutive 16-point tiles (amortizes weight gathers)
__global__ void __launch_bounds__(256)
k_attn(const float* __restrict__ feats, const int* __restrict__ bidx,
       const float* __restrict__ aW1, const float* __restrict__ ab1,
       const float* __restrict__ aW2, const float* __restrict__ ab2,
       float* __restrict__ raw, unsigned* __restrict__ maxenc, int n)
{
  __shared__ unsigned lmax[BATCH];
  if (threadIdx.x < BATCH) lmax[threadIdx.x] = 0u;
  __syncthreads();

  const int lane    = threadIdx.x & 31;
  const int lane_lo = lane & 15;
  const int lane_hi = lane >> 4;
  const int wave    = blockIdx.x * 8 + (threadIdx.x >> 5);

  // B operand: aW1 [K=64][N=32], B-layout elem j -> K = (j<8 ? j : j+8) + 8*lane_hi, N = lane_lo
  v16bf Bt[2][2];
#pragma unroll
  for (int kc = 0; kc < 2; ++kc)
#pragma unroll
    for (int nh = 0; nh < 2; ++nh)
#pragma unroll
      for (int j = 0; j < 16; ++j)
        Bt[kc][nh][j] = (__bf16)aW1[(kc * 32 + 8 * lane_hi + (j & 7) + ((j >> 3) << 4)) * 32 + nh * 16 + lane_lo];

  const float w2a = aW2[lane_lo], w2b = aW2[16 + lane_lo];
  const float b1a = ab1[lane_lo], b1b = ab1[16 + lane_lo];
  const float b2  = ab2[0];

#pragma unroll 1
  for (int t = 0; t < 4; ++t) {
    const int tile = wave * 4 + t;
    const int row0 = tile * 16;
    if (row0 >= n) break;                 // wave-uniform; EXEC stays full for WMMA

    v8f c0 = {}; v8f c1 = {};
    const int m  = row0 + lane_lo;
    const int mc = (m < n) ? m : (n - 1); // rows >= n compute garbage, never stored
#pragma unroll
    for (int kc = 0; kc < 2; ++kc) {
      // A operand: elem j -> K = (j<8 ? j : j+8) + 8*lane_hi (two aligned 8-float runs)
      const float* base = feats + (size_t)mc * CDIM + kc * 32 + 8 * lane_hi;
      f32x4 q0 = *(const f32x4*)(base);
      f32x4 q1 = *(const f32x4*)(base + 4);
      f32x4 q2 = *(const f32x4*)(base + 16);
      f32x4 q3 = *(const f32x4*)(base + 20);
      v16bf A;
#pragma unroll
      for (int j = 0; j < 4; ++j) {
        A[j]      = (__bf16)q0[j];
        A[4 + j]  = (__bf16)q1[j];
        A[8 + j]  = (__bf16)q2[j];
        A[12 + j] = (__bf16)q3[j];
      }
      c0 = __builtin_amdgcn_wmma_f32_16x16x32_bf16(false, A, false, Bt[kc][0], (short)0, c0, false, false);
      c1 = __builtin_amdgcn_wmma_f32_16x16x32_bf16(false, A, false, Bt[kc][1], (short)0, c1, false, false);
    }

    // bias + relu + dot with aW2; C layout: VGPR r -> M = r + 8*lane_hi, col = lane_lo (+16 for c1)
    float part[8];
#pragma unroll
    for (int r = 0; r < 8; ++r) {
      float h0 = fmaxf(c0[r] + b1a, 0.0f);
      float h1 = fmaxf(c1[r] + b1b, 0.0f);
      part[r] = h0 * w2a + h1 * w2b;
    }
#pragma unroll
    for (int off = 8; off >= 1; off >>= 1)
#pragma unroll
      for (int r = 0; r < 8; ++r)
        part[r] += __shfl_xor(part[r], off, 32);

    if (lane_lo == 0) {
#pragma unroll
      for (int r = 0; r < 8; ++r) {
        int mm = row0 + 8 * lane_hi + r;
        if (mm < n) {
          float v = part[r] + b2;
          raw[mm] = v;
          atomicMax(&lmax[bidx[mm]], fkey(v));
        }
      }
    }
  }
  __syncthreads();
  if (threadIdx.x < BATCH && lmax[threadIdx.x]) atomicMax(&maxenc[threadIdx.x], lmax[threadIdx.x]);
}

// ---------------- softmax: exp(raw - max) and per-batch sums ----------------
__global__ void __launch_bounds__(256)
k_expsum(const float* __restrict__ raw, const int* __restrict__ bidx,
         const unsigned* __restrict__ maxenc,
         float* __restrict__ sexp, float* __restrict__ ssum, int n)
{
  __shared__ float ls[BATCH];
  if (threadIdx.x < BATCH) ls[threadIdx.x] = 0.0f;
  __syncthreads();
  int i = blockIdx.x * blockDim.x + threadIdx.x;
  if (i < n) {
    int b = bidx[i];
    float e = __expf(raw[i] - fdec(maxenc[b]));
    sexp[i] = e;
    atomicAdd(&ls[b], e);
  }
  __syncthreads();
  if (threadIdx.x < BATCH && ls[threadIdx.x] != 0.0f) atomicAdd(&ssum[threadIdx.x], ls[threadIdx.x]);
}

// ---------------- weighted scatter into finest-level bins only --------------
__global__ void __launch_bounds__(256)
k_scatter(const float* __restrict__ feats, const int* __restrict__ bidx,
          const int* __restrict__ vox, const float* __restrict__ sexp,
          const float* __restrict__ ssum,
          float* __restrict__ fine, float* __restrict__ cnt, int n)
{
  int p = blockIdx.x * 4 + (threadIdx.x >> 6);
  int c = threadIdx.x & 63;
  if (p >= n) return;
  int b = bidx[p];
  float attn = sexp[p] / (ssum[b] + 1e-8f);
  int cx = vox[3 * p], cy = vox[3 * p + 1], cz = vox[3 * p + 2];
  int bin = b * FINEB + (cx >> 6) * 32 + (cy >> 6) * 8 + (cz >> 5);
  float w = feats[(size_t)p * CDIM + c] * attn;
  atomicAdd(&fine[(size_t)bin * CDIM + c], w);
  if (c == 0) atomicAdd(&cnt[bin], 1.0f);
}

// ---------------- hierarchical pooled means -> bf16 pyramid [64 x 12864] ----
__global__ void __launch_bounds__(64)
k_pyramid(const float* __restrict__ fine, const float* __restrict__ cnt,
          __bf16* __restrict__ pyr)
{
  const int b = blockIdx.x, c = threadIdx.x;
  __shared__ float cn3[FINEB];
  cn3[c] = cnt[b * FINEB + c];
  cn3[64 + c] = cnt[b * FINEB + 64 + c];
  __syncthreads();

  __bf16* row = pyr + (size_t)b * KDIM;
  float s1[8], n1[8];
#pragma unroll
  for (int i = 0; i < 8; ++i) { s1[i] = 0.0f; n1[i] = 0.0f; }
  float s0 = 0.0f, n0 = 0.0f;

#pragma unroll
  for (int z2 = 0; z2 < 64; ++z2) {          // level-2 (4,4,4) bin = fine>>1
    const int bx = z2 >> 4, by = (z2 >> 2) & 3, bz2 = z2 & 3;
    const int f0 = 2 * z2, f1 = f0 + 1;
    float a  = fine[(size_t)(b * FINEB + f0) * CDIM + c];
    float d  = fine[(size_t)(b * FINEB + f1) * CDIM + c];
    float na = cn3[f0], nd = cn3[f1];
    row[4672 + f0 * 64 + c] = (__bf16)(na > 0.0f ? a / na : 0.0f);   // level 3
    row[4672 + f1 * 64 + c] = (__bf16)(nd > 0.0f ? d / nd : 0.0f);
    float s2 = a + d, nn2 = na + nd;
    row[576 + z2 * 64 + c] = (__bf16)(nn2 > 0.0f ? s2 / nn2 : 0.0f); // level 2
    const int l1 = (bx >> 1) * 4 + (by >> 1) * 2 + (bz2 >> 1);
    s1[l1] += s2; n1[l1] += nn2;
  }
#pragma unroll
  for (int l1 = 0; l1 < 8; ++l1) {
    row[64 + l1 * 64 + c] = (__bf16)(n1[l1] > 0.0f ? s1[l1] / n1[l1] : 0.0f); // level 1
    s0 += s1[l1]; n0 += n1[l1];
  }
  row[c] = (__bf16)(n0 > 0.0f ? s0 / n0 : 0.0f);                     // level 0
}

// ---------------- GEMM1: hidden = relu(pyramid[64,12864] @ pW1 + pb1) -------
// 2 blocks x 1024 threads = 64 waves (4 M-tiles x 16 N-tiles).
// B chunks (32x256 bf16 = 16KB) staged in LDS via async-to-LDS, double-buffered.
__global__ void __launch_bounds__(1024)
k_gemm1(const __bf16* __restrict__ Abf, const __bf16* __restrict__ Wbf,
        const float* __restrict__ bias, __bf16* __restrict__ hid)
{
  const int tid     = threadIdx.x;
  const int lane    = tid & 31;
  const int lane_lo = lane & 15, lane_hi = lane >> 4;
  const int wid     = blockIdx.x * 32 + (tid >> 5);  // 0..63
  const int mt = wid >> 4, nt = wid & 15;
  const int ncol = nt * 16 + lane_lo;
  const int NC = KDIM / 32;                          // 402 chunks

  v8f acc = {};
  const __bf16* arow = Abf + (size_t)(mt * 16 + lane_lo) * KDIM + 8 * lane_hi;

#if HAVE_ASYNC_LDS
  __shared__ __bf16 Bs[2][32 * HID];                 // 2 x 16KB
  auto stage = [&](int kc, int buf) {
    const void* g = (const char*)(Wbf + (size_t)kc * 32 * HID) + (size_t)tid * 16;
    void* l = (char*)(&Bs[buf][0]) + (size_t)tid * 16;
    __builtin_amdgcn_global_load_async_to_lds_b128((as1_v4i)g, (as3_v4i)l, 0, 0);
  };
  auto compute = [&](int kc) {
    bf16x8 a0 = *(const bf16x8*)(arow + kc * 32);
    bf16x8 a1 = *(const bf16x8*)(arow + kc * 32 + 16);
    v16bf A = __builtin_shufflevector(a0, a1, 0,1,2,3,4,5,6,7,8,9,10,11,12,13,14,15);
    const __bf16* Bl = &Bs[kc & 1][0];
    v16bf Bv;
#pragma unroll
    for (int j = 0; j < 16; ++j)
      Bv[j] = Bl[(8 * lane_hi + (j & 7) + ((j >> 3) << 4)) * HID + ncol];
    acc = __builtin_amdgcn_wmma_f32_16x16x32_bf16(false, A, false, Bv, (short)0, acc, false, false);
  };

  stage(0, 0);
  for (int kc = 0; kc < NC - 1; ++kc) {
    stage(kc + 1, (kc + 1) & 1);
    __builtin_amdgcn_s_wait_asynccnt(1);   // chunk kc resident
    __syncthreads();
    compute(kc);
    __syncthreads();                       // all waves done before buffer reuse
  }
  __builtin_amdgcn_s_wait_asynccnt(0);
  __syncthreads();
  compute(NC - 1);
#else
  for (int kc = 0; kc < NC; ++kc) {
    __builtin_prefetch(arow + (kc + 8) * 32, 0, 0);
    __builtin_prefetch(Wbf + (size_t)((kc + 2) * 32 + 16 * lane_hi) * HID + ncol, 0, 0);
    bf16x8 a0 = *(const bf16x8*)(arow + kc * 32);
    bf16x8 a1 = *(const bf16x8*)(arow + kc * 32 + 16);
    v16bf A = __builtin_shufflevector(a0, a1, 0,1,2,3,4,5,6,7,8,9,10,11,12,13,14,15);
    v16bf Bv;
#pragma unroll
    for (int j = 0; j < 16; ++j)
      Bv[j] = Wbf[(size_t)(kc * 32 + 8 * lane_hi + (j & 7) + ((j >> 3) << 4)) * HID + ncol];
    acc = __builtin_amdgcn_wmma_f32_16x16x32_bf16(false, A, false, Bv, (short)0, acc, false, false);
  }
#endif

  const float bn = bias[ncol];
#pragma unroll
  for (int r = 0; r < 8; ++r) {
    int m = mt * 16 + 8 * lane_hi + r;
    hid[(size_t)m * HID + ncol] = (__bf16)fmaxf(acc[r] + bn, 0.0f);
  }
}

// ---------------- GEMM2: out = hidden[64,256] @ pW2 + pb2 -------------------
__global__ void __launch_bounds__(256)
k_gemm2(const __bf16* __restrict__ Abf, const __bf16* __restrict__ Wbf,
        const float* __restrict__ bias, float* __restrict__ out)
{
  const int lane    = threadIdx.x & 31;
  const int lane_lo = lane & 15, lane_hi = lane >> 4;
  const int wid     = blockIdx.x * 8 + (threadIdx.x >> 5);
  const int mt = wid >> 4, nt = wid & 15;
  const int ncol = nt * 16 + lane_lo;

  v8f acc = {};
  const __bf16* arow = Abf + (size_t)(mt * 16 + lane_lo) * HID + 8 * lane_hi;
#pragma unroll
  for (int kc = 0; kc < HID / 32; ++kc) {
    bf16x8 a0 = *(const bf16x8*)(arow + kc * 32);
    bf16x8 a1 = *(const bf16x8*)(arow + kc * 32 + 16);
    v16bf A = __builtin_shufflevector(a0, a1, 0,1,2,3,4,5,6,7,8,9,10,11,12,13,14,15);
    v16bf Bv;
#pragma unroll
    for (int j = 0; j < 16; ++j)
      Bv[j] = Wbf[(size_t)(kc * 32 + 8 * lane_hi + (j & 7) + ((j >> 3) << 4)) * HID + ncol];
    acc = __builtin_amdgcn_wmma_f32_16x16x32_bf16(false, A, false, Bv, (short)0, acc, false, false);
  }
  const float bn = bias[ncol];
#pragma unroll
  for (int r = 0; r < 8; ++r) {
    int m = mt * 16 + 8 * lane_hi + r;
    out[(size_t)m * HID + ncol] = acc[r] + bn;
  }
}

extern "C" void kernel_launch(void* const* d_in, const int* in_sizes, int n_in,
                              void* d_out, int out_size, void* d_ws, size_t ws_size,
                              hipStream_t stream)
{
  const float* feats = (const float*)d_in[0];
  const int*   bidx  = (const int*)d_in[1];
  const int*   vox   = (const int*)d_in[2];
  // d_in[3] = batch_size scalar (device); B=64 per reference
  const float* aW1 = (const float*)d_in[4];
  const float* ab1 = (const float*)d_in[5];
  const float* aW2 = (const float*)d_in[6];
  const float* ab2 = (const float*)d_in[7];
  const float* pW1 = (const float*)d_in[8];
  const float* pb1 = (const float*)d_in[9];
  const float* pW2 = (const float*)d_in[10];
  const float* pb2 = (const float*)d_in[11];
  const int n = in_sizes[1];   // N points

  // workspace layout (accumulator block first so one zero-pass covers it)
  char* ws = (char*)d_ws;
  unsigned* maxenc = (unsigned*)(ws + 0);                     //   64 u32
  float*    ssum   = (float*)(ws + 256);                      //   64 f32
  float*    fine   = (float*)(ws + 512);                      // 8192*64 f32
  float*    cnt    = (float*)(ws + 512 + (size_t)BATCH * FINEB * CDIM * 4);
  size_t off = 512 + (size_t)BATCH * FINEB * CDIM * 4 + (size_t)BATCH * FINEB * 4; // 2,130,432
  const int zero_words = (int)(off / 4);
  float*  raw   = (float*)(ws + off); off += (((size_t)n * 4) + 255) & ~(size_t)255;
  float*  sexp  = (float*)(ws + off); off += (((size_t)n * 4) + 255) & ~(size_t)255;
  __bf16* pyrbf = (__bf16*)(ws + off); off += (size_t)BATCH * KDIM * 2;
  __bf16* pW1bf = (__bf16*)(ws + off); off += (size_t)KDIM * HID * 2;
  __bf16* pW2bf = (__bf16*)(ws + off); off += (size_t)HID * HID * 2;
  __bf16* hidbf = (__bf16*)(ws + off); off += (size_t)BATCH * HID * 2;

  k_zero<<<(zero_words + 255) / 256, 256, 0, stream>>>((unsigned*)ws, zero_words);

  const int ncvt = KDIM * HID;
  k_cvt<<<(ncvt + 255) / 256, 256, 0, stream>>>(pW1, pW2, pW1bf, pW2bf, ncvt, HID * HID);

  const int tiles = (n + 15) / 16;
  const int waves = (tiles + 3) / 4;
  k_attn<<<(waves + 7) / 8, 256, 0, stream>>>(feats, bidx, aW1, ab1, aW2, ab2, raw, maxenc, n);

  k_expsum<<<(n + 255) / 256, 256, 0, stream>>>(raw, bidx, maxenc, sexp, ssum, n);

  k_scatter<<<(n + 3) / 4, 256, 0, stream>>>(feats, bidx, vox, sexp, ssum, fine, cnt, n);

  k_pyramid<<<BATCH, 64, 0, stream>>>(fine, cnt, pyrbf);

  k_gemm1<<<2, 1024, 0, stream>>>(pyrbf, pW1bf, pb1, hidbf);
  k_gemm2<<<8, 256, 0, stream>>>(hidbf, pW2bf, pb2, (float*)d_out);

  (void)n_in; (void)out_size; (void)ws_size;
}
